// Attention_81827716923787
// MI455X (gfx1250) — matrix operand order repaired and checked by reading, not compiled, not run on hardware
//
#include <hip/hip_runtime.h>

// ---------------------------------------------------------------------------
// MI455X (gfx1250, wave32) sliding-window attention with RoPE, bf16 WMMA path.
// D = A*B + C via v_wmma_f32_16x16x32_bf16 (16x16 tiles, K=32).
// Global->LDS staging of straight-copy tiles uses GLOBAL_LOAD_ASYNC_TO_LDS_B128
// (ASYNCcnt) per CDNA5 ISA.
// ---------------------------------------------------------------------------

typedef __bf16 bf16;
typedef __attribute__((ext_vector_type(8)))  __bf16 v8bf;
typedef __attribute__((ext_vector_type(16))) __bf16 v16bf;
typedef __attribute__((ext_vector_type(8)))  float  v8f;

// Problem constants (from the reference)
constexpr int B_    = 2;
constexpr int S_    = 2048;
constexpr int DIM_  = 2048;
constexpr int H_    = 16;
constexpr int KVH_  = 4;
constexpr int HD_   = 128;
constexpr int HALF_ = HD_ / 2;       // 64
constexpr int HDE_  = 3 * HALF_;     // 192 extended rope dim: [qc-qs, qc+qs, qi]
constexpr int WIN_  = 1024;
#define SCALE_F 0.08838834764831845f // 128^-0.5

__device__ __forceinline__ v8f vzero8() {
  v8f z;
#pragma unroll
  for (int i = 0; i < 8; ++i) z[i] = 0.0f;
  return z;
}

__device__ __forceinline__ v16bf cat16(v8bf lo, v8bf hi) {
  return __builtin_shufflevector(lo, hi, 0,1,2,3,4,5,6,7,8,9,10,11,12,13,14,15);
}

__device__ __forceinline__ v8f wmma_bf16(v16bf a, v16bf b, v8f c) {
  // (neg_a, A, neg_b, B, c_mod, C, reuse_a, reuse_b)
  return __builtin_amdgcn_wmma_f32_16x16x32_bf16(false, a, false, b, (short)0, c,
                                                 false, false);
}

// Async global -> LDS copy of 16 bytes per lane (per-wave instruction).
__device__ __forceinline__ void async_copy_b128(const bf16* gaddr, bf16* lds_ptr) {
  unsigned lds = (unsigned)(size_t)lds_ptr;  // generic LDS addr[31:0] = LDS offset
  unsigned long long ga = (unsigned long long)gaddr;
  asm volatile("global_load_async_to_lds_b128 %0, %1, off"
               :: "v"(lds), "v"(ga) : "memory");
}
__device__ __forceinline__ void async_wait0() {
  asm volatile("s_wait_asynccnt 0" ::: "memory");
}

__device__ __forceinline__ float rmax16(float v) {
#pragma unroll
  for (int m = 1; m < 16; m <<= 1) v = fmaxf(v, __shfl_xor(v, m, 32));
  return v;
}
__device__ __forceinline__ float rsum16(float v) {
#pragma unroll
  for (int m = 1; m < 16; m <<= 1) v += __shfl_xor(v, m, 32);
  return v;
}

// ---------------------------------------------------------------------------
// Elementwise f32 -> bf16 convert
// ---------------------------------------------------------------------------
__global__ void cvt_bf16_kernel(const float* __restrict__ in, bf16* __restrict__ out, int n) {
  int i = blockIdx.x * 256 + threadIdx.x;
  if (i < n) out[i] = (bf16)in[i];
}

// ---------------------------------------------------------------------------
// Tiled bf16 WMMA GEMM:  C[M,N] = A[M,K] * W[K,N]
//   block = 256 threads (8 waves); tile 256(M) x 64(N); K chunk = 32.
//   Wave owns 32 rows (2 A-frags, 8 accumulators) -> 8 WMMAs per chunk.
//   A staged [m][k] row-major in LDS via async global->LDS; W staged
//   transposed [n][k] so B-fragments are two contiguous ds_read_b128.
// ---------------------------------------------------------------------------
template <int OUTF32>
__global__ __launch_bounds__(256) void gemm_bf16_kernel(
    const bf16* __restrict__ A, const bf16* __restrict__ W, void* __restrict__ Cout,
    int M, int N, int K) {
  constexpr int TM = 256, TN = 64, TK = 32;
  __shared__ __align__(16) bf16 As[TM * TK];   // 16 KB
  __shared__ __align__(16) bf16 BsT[TN * TK];  //  4 KB, [n][k]

  const int t    = threadIdx.x;
  const int lane = t & 31;
  const int w    = t >> 5;
  const int hi   = (lane >> 4) & 1;
  const int ln   = lane & 15;
  const int m0   = blockIdx.x * TM;
  const int n0   = blockIdx.y * TN;

  v8f acc[2][4];
#pragma unroll
  for (int i = 0; i < 2; ++i)
#pragma unroll
    for (int j = 0; j < 4; ++j) acc[i][j] = vzero8();

  const int bkS = t >> 3;        // B stage: k 0..31
  const int bnS = (t & 7) * 8;   // B stage: n sub-offset

  for (int kc = 0; kc < K; kc += TK) {
    __syncthreads();
    // ---- stage A tile via async global->LDS (1024 x b128, 4/thread) ----
#pragma unroll
    for (int i = 0; i < 4; ++i) {
      const int vecid = t + 256 * i;
      const int row = vecid >> 2;
      const int ko  = (vecid & 3) * 8;
      async_copy_b128(A + (size_t)(m0 + row) * K + kc + ko, &As[row * TK + ko]);
    }
    // ---- stage W tile transposed into [n][k] ----
    {
      const bf16* g = W + (size_t)(kc + bkS) * N + n0 + bnS;
      v8bf vv = *(const v8bf*)g;
      if (kc + TK < K) __builtin_prefetch(g + (size_t)TK * N, 0, 1);
#pragma unroll
      for (int i = 0; i < 8; ++i) BsT[(bnS + i) * TK + bkS] = vv[i];
    }
    async_wait0();
    __syncthreads();

    // ---- A fragments: rows w*32 + {0..15, 16..31} ----
    v16bf afrag[2];
#pragma unroll
    for (int mt = 0; mt < 2; ++mt) {
      const int arow = w * 32 + mt * 16 + ln;
      v8bf a_lo = *(const v8bf*)&As[arow * TK + hi * 8];
      v8bf a_hi = *(const v8bf*)&As[arow * TK + 16 + hi * 8];
      afrag[mt] = cat16(a_lo, a_hi);
    }

#pragma unroll
    for (int nt = 0; nt < 4; ++nt) {
      const int bn = nt * 16 + ln;
      v8bf b_lo = *(const v8bf*)&BsT[bn * TK + hi * 8];
      v8bf b_hi = *(const v8bf*)&BsT[bn * TK + 16 + hi * 8];
      v16bf bfrag = cat16(b_lo, b_hi);
#pragma unroll
      for (int mt = 0; mt < 2; ++mt)
        acc[mt][nt] = wmma_bf16(afrag[mt], bfrag, acc[mt][nt]);
    }
  }

  // ---- epilogue: C layout = col(lane&15), rows v + hi*8 ----
#pragma unroll
  for (int mt = 0; mt < 2; ++mt) {
#pragma unroll
    for (int nt = 0; nt < 4; ++nt) {
      const int col = n0 + nt * 16 + ln;
#pragma unroll
      for (int v = 0; v < 8; ++v) {
        const int r = m0 + w * 32 + mt * 16 + hi * 8 + v;
        const float val = acc[mt][nt][v];
        if (OUTF32) ((float*)Cout)[(size_t)r * N + col] = val;
        else        ((bf16*)Cout)[(size_t)r * N + col]  = (bf16)val;
      }
    }
  }
}

// ---------------------------------------------------------------------------
// RoPE extension per reference:
//   qe = [qr*(c-s), qr*(c+s), qi]  (192 dims), layout (B,H,S,HDE)
//   ke likewise, layout (B,KVH,S,HDE)
// ---------------------------------------------------------------------------
__global__ void rope_kernel(const bf16* __restrict__ qraw, const bf16* __restrict__ kraw,
                            const float* __restrict__ cosb, const float* __restrict__ sinb,
                            bf16* __restrict__ qe, bf16* __restrict__ ke) {
  const int NQ = B_ * S_ * H_ * HALF_;
  const int NK = B_ * S_ * KVH_ * HALF_;
  int gid = blockIdx.x * 256 + threadIdx.x;
  if (gid < NQ) {
    int i = gid % HALF_; int r = gid / HALF_;
    int h = r % H_;      r /= H_;
    int s = r % S_;      int b = r / S_;
    float c  = cosb[s * HALF_ + i];
    float sn = sinb[s * HALF_ + i];
    const bf16* src = qraw + ((size_t)(b * S_ + s)) * (H_ * HD_) + h * HD_;
    float xr = (float)src[i];
    float xi = (float)src[HALF_ + i];
    bf16* dst = qe + ((size_t)(b * H_ + h) * S_ + s) * HDE_;
    dst[i]             = (bf16)(xr * (c - sn));
    dst[HALF_ + i]     = (bf16)(xr * (c + sn));
    dst[2 * HALF_ + i] = (bf16)xi;
  } else if (gid < NQ + NK) {
    int g = gid - NQ;
    int i = g % HALF_; int r = g / HALF_;
    int h = r % KVH_;  r /= KVH_;
    int s = r % S_;    int b = r / S_;
    float c  = cosb[s * HALF_ + i];
    float sn = sinb[s * HALF_ + i];
    const bf16* src = kraw + ((size_t)(b * S_ + s)) * (KVH_ * HD_) + h * HD_;
    float xr = (float)src[i];
    float xi = (float)src[HALF_ + i];
    bf16* dst = ke + ((size_t)(b * KVH_ + h) * S_ + s) * HDE_;
    dst[i]             = (bf16)(xr * (c - sn));
    dst[HALF_ + i]     = (bf16)(xr * (c + sn));
    dst[2 * HALF_ + i] = (bf16)xi;
  }
}

// V layout permute: (b,s,kvh,d) -> (b,kvh,s,d)
__global__ void vperm_kernel(const bf16* __restrict__ vraw, bf16* __restrict__ vp) {
  const int N = B_ * S_ * KVH_ * HD_;
  int gid = blockIdx.x * 256 + threadIdx.x;
  if (gid >= N) return;
  int d = gid % HD_; int r = gid / HD_;
  int h = r % KVH_;  r /= KVH_;
  int s = r % S_;    int b = r / S_;
  vp[((size_t)(b * KVH_ + h) * S_ + s) * HD_ + d] =
      vraw[((size_t)(b * S_ + s)) * (KVH_ * HD_) + h * HD_ + d];
}

// ---------------------------------------------------------------------------
// Sliding-window flash attention, bf16 WMMA.
//   grid = (S/128, H, B); block = 256 = 8 waves; wave w owns 16 q rows.
//   Key loop over 32-key tiles in the union window of the 128 q rows.
//   k~ tile staged via async global->LDS; V tile transposed through VGPRs.
// ---------------------------------------------------------------------------
__global__ __launch_bounds__(256) void attn_kernel(
    const bf16* __restrict__ qe, const bf16* __restrict__ ke,
    const bf16* __restrict__ vp, bf16* __restrict__ attn_out) {
  __shared__ __align__(16) bf16 Kt[32 * HDE_];    // [key][e]   12 KB
  __shared__ __align__(16) bf16 VtT[HD_ * 32];    // [d][key]    8 KB
  __shared__ __align__(16) bf16 Pb[8 * 16 * 32];  // per-wave P  8 KB

  const int t    = threadIdx.x;
  const int lane = t & 31;
  const int w    = t >> 5;
  const int hi   = (lane >> 4) & 1;
  const int ln   = lane & 15;
  const int b    = blockIdx.z;
  const int h    = blockIdx.y;
  const int Q0   = blockIdx.x * 128;
  const int hk   = h / (H_ / KVH_);

  // ---- q~ A-fragments for the 6 K-chunks of HDE=192 ----
  const bf16* qbase = qe + ((size_t)(b * H_ + h) * S_ + (Q0 + w * 16 + ln)) * HDE_;
  v16bf qa[6];
#pragma unroll
  for (int c = 0; c < 6; ++c) {
    v8bf lo = *(const v8bf*)(qbase + c * 32 + hi * 8);
    v8bf hp = *(const v8bf*)(qbase + c * 32 + 16 + hi * 8);
    qa[c] = cat16(lo, hp);
  }

  v8f O[8];
  float m[8], l[8];
#pragma unroll
  for (int i = 0; i < 8; ++i) { O[i] = vzero8(); m[i] = -3.0e38f; l[i] = 0.0f; }

  const bf16* kbase = ke + ((size_t)(b * KVH_ + hk) * S_) * HDE_;
  const bf16* vbase = vp + ((size_t)(b * KVH_ + hk) * S_) * HD_;

  int kbeg = Q0 - (WIN_ - 1); if (kbeg < 0) kbeg = 0;
  kbeg &= ~31;
  const int kend = Q0 + 128;

  for (int kt = kbeg; kt < kend; kt += 32) {
    __syncthreads();
    // ---- stage k~ tile [32][192] via async (768 x 16B vectors, 3/thread) ----
#pragma unroll
    for (int i = 0; i < 3; ++i) {
      int vecid = t + 256 * i;
      int key = vecid / 24, eo = (vecid % 24) * 8;
      async_copy_b128(kbase + (size_t)(kt + key) * HDE_ + eo, &Kt[key * HDE_ + eo]);
    }
    // ---- stage V tile transposed [d][key] (overlaps with async k~) ----
#pragma unroll
    for (int i = 0; i < 2; ++i) {
      int vecid = t + 256 * i;
      int key = vecid / 16, doff = (vecid % 16) * 8;
      v8bf vv = *(const v8bf*)(vbase + (size_t)(kt + key) * HD_ + doff);
#pragma unroll
      for (int j = 0; j < 8; ++j) VtT[(doff + j) * 32 + key] = vv[j];
    }
    async_wait0();
    __syncthreads();

    // ---- scores: two 16x16 tiles (keys kt..kt+15, kt+16..kt+31) ----
    v8f s0 = vzero8(), s1 = vzero8();
#pragma unroll
    for (int c = 0; c < 6; ++c) {
      {
        const bf16* p = &Kt[ln * HDE_ + c * 32 + hi * 8];
        s0 = wmma_bf16(qa[c], cat16(*(const v8bf*)p, *(const v8bf*)(p + 16)), s0);
      }
      {
        const bf16* p = &Kt[(16 + ln) * HDE_ + c * 32 + hi * 8];
        s1 = wmma_bf16(qa[c], cat16(*(const v8bf*)p, *(const v8bf*)(p + 16)), s1);
      }
    }

    // ---- mask + online softmax (row reductions inside 16-lane halves) ----
    const int k0 = kt + ln;
    const int k1 = kt + 16 + ln;
#pragma unroll
    for (int v = 0; v < 8; ++v) {
      const int r = Q0 + w * 16 + hi * 8 + v;
      float x0 = s0[v] * SCALE_F + ((k0 <= r && k0 > r - WIN_) ? 0.0f : -1.0e9f);
      float x1 = s1[v] * SCALE_F + ((k1 <= r && k1 > r - WIN_) ? 0.0f : -1.0e9f);
      float tm = rmax16(fmaxf(x0, x1));
      float nm = fmaxf(m[v], tm);
      float sf = __expf(m[v] - nm);
      float p0 = __expf(x0 - nm);
      float p1 = __expf(x1 - nm);
      l[v] = l[v] * sf + rsum16(p0 + p1);
      m[v] = nm;
#pragma unroll
      for (int nt = 0; nt < 8; ++nt) O[nt][v] *= sf;
      const int rl = hi * 8 + v;
      Pb[w * 512 + rl * 32 + ln]      = (bf16)p0;
      Pb[w * 512 + rl * 32 + 16 + ln] = (bf16)p1;
    }
    asm volatile("s_wait_dscnt 0" ::: "memory");  // P store -> P frag read (same wave)

    // ---- P A-fragment (16 rows x 32 keys) ----
    const bf16* pp = &Pb[w * 512 + ln * 32];
    v16bf pf = cat16(*(const v8bf*)(pp + hi * 8), *(const v8bf*)(pp + 16 + hi * 8));

    // ---- O += P * V over the 8 16-wide d tiles ----
#pragma unroll
    for (int nt = 0; nt < 8; ++nt) {
      const bf16* vb = &VtT[(nt * 16 + ln) * 32 + hi * 8];
      O[nt] = wmma_bf16(pf, cat16(*(const v8bf*)vb, *(const v8bf*)(vb + 16)), O[nt]);
    }
  }

  // ---- normalize + store to (b,s,h,d) bf16 ----
#pragma unroll
  for (int nt = 0; nt < 8; ++nt) {
#pragma unroll
    for (int v = 0; v < 8; ++v) {
      const int r = Q0 + w * 16 + hi * 8 + v;
      float val = O[nt][v] / l[v];
      attn_out[((size_t)(b * S_ + r) * H_ + h) * HD_ + nt * 16 + ln] = (bf16)val;
    }
  }
}

// ---------------------------------------------------------------------------
// Host-side orchestration
// ---------------------------------------------------------------------------
extern "C" void kernel_launch(void* const* d_in, const int* in_sizes, int n_in,
                              void* d_out, int out_size, void* d_ws, size_t ws_size,
                              hipStream_t stream) {
  (void)in_sizes; (void)n_in; (void)out_size; (void)ws_size;
  const float* x    = (const float*)d_in[0];
  const float* cosb = (const float*)d_in[1];
  const float* sinb = (const float*)d_in[2];
  // d_in[3] = positions, d_in[4] = mask: window logic is reproduced in-kernel.
  const float* wq = (const float*)d_in[5];
  const float* wk = (const float*)d_in[6];
  const float* wv = (const float*)d_in[7];
  const float* wo = (const float*)d_in[8];
  float* out = (float*)d_out;

  const int ROWS = B_ * S_;                 // 4096
  const int NX  = ROWS * DIM_;              // 8,388,608
  const int NWQ = DIM_ * H_ * HD_;          // 4,194,304
  const int NWK = DIM_ * KVH_ * HD_;        // 1,048,576
  const int NWV = NWK;
  const int NWO = H_ * HD_ * DIM_;          // 4,194,304
  const int NK_RAW = ROWS * KVH_ * HD_;     // 2,097,152
  const int NQE = B_ * H_ * S_ * HDE_;      // 12,582,912
  const int NKE = B_ * KVH_ * S_ * HDE_;    // 3,145,728
  const int NVP = B_ * KVH_ * S_ * HD_;     // 2,097,152

  char* p = (char*)d_ws;
  auto alloc = [&](size_t elems) {
    bf16* r = (bf16*)p;
    p += (elems * sizeof(bf16) + 255) & ~(size_t)255;
    return r;
  };
  bf16* xb   = alloc(NX);
  bf16* wqb  = alloc(NWQ);
  bf16* wkb  = alloc(NWK);
  bf16* wvb  = alloc(NWV);
  bf16* wob  = alloc(NWO);
  bf16* qraw = alloc(NX);        // 4096 x 2048
  bf16* kraw = alloc(NK_RAW);    // 4096 x 512
  bf16* vraw = alloc(NK_RAW);
  bf16* qe   = alloc(NQE);
  bf16* ke   = alloc(NKE);
  bf16* vp   = alloc(NVP);
  bf16* attn = alloc(NX);        // 4096 x 2048

  auto cvt = [&](const float* src, bf16* dst, int n) {
    cvt_bf16_kernel<<<(n + 255) / 256, 256, 0, stream>>>(src, dst, n);
  };
  cvt(x,  xb,  NX);
  cvt(wq, wqb, NWQ);
  cvt(wk, wkb, NWK);
  cvt(wv, wvb, NWV);
  cvt(wo, wob, NWO);

  // Projections (bf16 WMMA GEMMs)
  gemm_bf16_kernel<0><<<dim3(ROWS / 256, (H_ * HD_) / 64), 256, 0, stream>>>(
      xb, wqb, qraw, ROWS, H_ * HD_, DIM_);
  gemm_bf16_kernel<0><<<dim3(ROWS / 256, (KVH_ * HD_) / 64), 256, 0, stream>>>(
      xb, wkb, kraw, ROWS, KVH_ * HD_, DIM_);
  gemm_bf16_kernel<0><<<dim3(ROWS / 256, (KVH_ * HD_) / 64), 256, 0, stream>>>(
      xb, wvb, vraw, ROWS, KVH_ * HD_, DIM_);

  // RoPE extension + V permute
  {
    const int n = B_ * S_ * H_ * HALF_ + B_ * S_ * KVH_ * HALF_;
    rope_kernel<<<(n + 255) / 256, 256, 0, stream>>>(qraw, kraw, cosb, sinb, qe, ke);
    vperm_kernel<<<(NVP + 255) / 256, 256, 0, stream>>>(vraw, vp);
  }

  // Flash attention
  attn_kernel<<<dim3(S_ / 128, H_, B_), 256, 0, stream>>>(qe, ke, vp, attn);

  // Output projection -> fp32 d_out
  gemm_bf16_kernel<1><<<dim3(ROWS / 256, DIM_ / 64), 256, 0, stream>>>(
      attn, wob, out, ROWS, DIM_, DIM_);
}
